// MultiheadSelfAttention_30502857736558
// MI455X (gfx1250) — matrix operand order repaired and checked by reading, not compiled, not run on hardware
//
#include <hip/hip_runtime.h>
#include <stdint.h>

#ifndef __has_builtin
#define __has_builtin(x) 0
#endif

// Problem constants (match reference)
#define BB 4
#define NN 2048
#define DD 1024
#define HH 16
#define DK 64

typedef __bf16 bf16;
typedef bf16  v16bf __attribute__((ext_vector_type(16)));
typedef float v8f   __attribute__((ext_vector_type(8)));
typedef int   v4i   __attribute__((vector_size(16)));   // builtin's pointee type

union V16 { v16bf v; uint32_t u[8]; };

__device__ __forceinline__ uint16_t f2b(float f) {
  union { float f; uint32_t u; } c; c.f = f;
  uint32_t r = c.u + 0x7FFFu + ((c.u >> 16) & 1u);   // round-to-nearest-even
  return (uint16_t)(r >> 16);
}

// ---- async global->LDS copy (ASYNCcnt path) with sync fallback -------------
#if __has_builtin(__builtin_amdgcn_global_load_async_to_lds_b128)
#define HAVE_ASYNC_LDS 1
#endif

__device__ __forceinline__ void async_copy16(const uint16_t* g, uint16_t* l) {
#ifdef HAVE_ASYNC_LDS
  __builtin_amdgcn_global_load_async_to_lds_b128((v4i*)g, (v4i*)l, 0, 0);
#else
  *(uint4*)l = *(const uint4*)g;
#endif
}

__device__ __forceinline__ void async_wait0() {
#ifdef HAVE_ASYNC_LDS
#if __has_builtin(__builtin_amdgcn_s_wait_asynccnt)
  __builtin_amdgcn_s_wait_asynccnt(0);
#else
  asm volatile("s_wait_asynccnt 0" ::: "memory");
#endif
#endif
}

// Max-reduction over each 16-lane row group via DPP (pure VALU).
#if __has_builtin(__builtin_amdgcn_update_dpp)
#define ROWSTEP_MAX(v, ctrl)                                                 \
  v = fmaxf(v, __int_as_float(__builtin_amdgcn_update_dpp(                   \
                 0, __float_as_int(v), ctrl, 0xF, 0xF, true)))
__device__ __forceinline__ float rowmax16(float v) {
  ROWSTEP_MAX(v, 0xB1);   // quad_perm(1,0,3,2)
  ROWSTEP_MAX(v, 0x4E);   // quad_perm(2,3,0,1)
  ROWSTEP_MAX(v, 0x124);  // row_ror:4
  ROWSTEP_MAX(v, 0x128);  // row_ror:8
  return v;
}
#else
__device__ __forceinline__ float rowmax16(float v) {
#pragma unroll
  for (int off = 1; off < 16; off <<= 1) v = fmaxf(v, __shfl_xor(v, off, 32));
  return v;
}
#endif

// ---------------------------------------------------------------- convert ---
__global__ void cvt_f32_bf16(const float* __restrict__ in,
                             uint16_t* __restrict__ out, int n) {
  int i = blockIdx.x * blockDim.x + threadIdx.x;
  int stride = gridDim.x * blockDim.x;
  for (; i < n; i += stride) out[i] = f2b(in[i]);
}

// ------------------------------------------------------------- bf16 GEMM ---
// C[M,Nn] = A[M,K] * B[K,Nn]; bf16 row-major, f32 accumulate.
// Block tile 128x256, BK=32, 8 waves (2x4), wave tile 64x64.
// Double-buffered LDS: stage tile i+1 while computing tile i (1 barrier/step).
template <bool OUT_BF16>
__global__ __launch_bounds__(256) void gemm_bf16(
    const uint16_t* __restrict__ A, const uint16_t* __restrict__ Bm,
    void* __restrict__ Cout, int M, int Nn, int K) {
  __shared__ __align__(16) uint16_t As[2][128 * 40];   // [m][k], pad 40
  __shared__ __align__(16) uint16_t Bs[2][256 * 40];   // transposed [n][k]

  const int tid   = threadIdx.x;
  const int lane  = tid & 31;
  const int wid   = tid >> 5;
  const int wm    = wid >> 2;          // 0..1  (64 rows)
  const int wn    = wid & 3;           // 0..3  (64 cols)
  const int laneN = lane & 15;
  const int half  = lane >> 4;
  const int m0 = blockIdx.y * 128;
  const int n0 = blockIdx.x * 256;

  auto stageA = [&](int k0, int bi) {
#pragma unroll
    for (int i = 0; i < 2; i++) {
      int idx = tid + 256 * i;               // 0..511
      int row = idx >> 2, chunk = idx & 3;
      async_copy16(A + (size_t)(m0 + row) * K + k0 + chunk * 8,
                   &As[bi][row * 40 + chunk * 8]);
    }
  };
  auto stageB = [&](int k0, int bi) {
#pragma unroll
    for (int i = 0; i < 2; i++) {
      int idx = tid + 256 * i;               // 0..511
      int kp = idx >> 5, seg = idx & 31;     // k-pair 0..15, n-seg of 8
      const uint16_t* g0 = Bm + (size_t)(k0 + 2 * kp) * Nn + n0 + seg * 8;
      uint4 a4 = *(const uint4*)g0;
      uint4 b4 = *(const uint4*)(g0 + Nn);
      const uint32_t* aw = (const uint32_t*)&a4;
      const uint32_t* bw = (const uint32_t*)&b4;
#pragma unroll
      for (int j = 0; j < 8; j++) {
        uint32_t av = (aw[j >> 1] >> ((j & 1) * 16)) & 0xFFFFu;
        uint32_t bv = (bw[j >> 1] >> ((j & 1) * 16)) & 0xFFFFu;
        *(uint32_t*)&Bs[bi][(seg * 8 + j) * 40 + 2 * kp] = av | (bv << 16);
      }
    }
  };

  v8f acc[4][4];
#pragma unroll
  for (int mi = 0; mi < 4; mi++)
#pragma unroll
    for (int ni = 0; ni < 4; ni++)
#pragma unroll
      for (int r = 0; r < 8; r++) acc[mi][ni][r] = 0.f;

  const int steps = K / 32;
  stageA(0, 0);
  stageB(0, 0);

  for (int it = 0; it < steps; ++it) {
    const int cur = it & 1;
    async_wait0();
    __syncthreads();
    if (it + 1 < steps) {
      stageA((it + 1) * 32, cur ^ 1);
      stageB((it + 1) * 32, cur ^ 1);
      if (it + 2 < steps) {
        int kn = (it + 2) * 32;
        __builtin_prefetch(A + (size_t)(m0 + (tid >> 1)) * K + kn + (tid & 1) * 16, 0, 1);
        __builtin_prefetch(Bm + (size_t)(kn + (tid >> 3)) * Nn + n0 + (tid & 7) * 32, 0, 1);
      }
    }

    // Load ALL fragments first (distinct registers -> overlapped ds waits)
    V16 a[4], b[4];
#pragma unroll
    for (int mi = 0; mi < 4; mi++) {
      int row = wm * 64 + mi * 16 + laneN;
#pragma unroll
      for (int j = 0; j < 8; j++) {
        int kk = 2 * j + (j >= 4 ? 8 : 0) + 8 * half;   // A 16x32 layout
        a[mi].u[j] = *(const uint32_t*)&As[cur][row * 40 + kk];
      }
    }
#pragma unroll
    for (int ni = 0; ni < 4; ni++) {
      int col = wn * 64 + ni * 16 + laneN;
#pragma unroll
      for (int j = 0; j < 8; j++) {
        int kk = 2 * j + (j >= 4 ? 8 : 0) + 8 * half;   // B 32x16 layout
        b[ni].u[j] = *(const uint32_t*)&Bs[cur][col * 40 + kk];
      }
    }
#pragma unroll
    for (int mi = 0; mi < 4; mi++)
#pragma unroll
      for (int ni = 0; ni < 4; ni++)
        acc[mi][ni] = __builtin_amdgcn_wmma_f32_16x16x32_bf16(
            false, a[mi].v, false, b[ni].v, (short)0, acc[mi][ni], false, false);
  }

  // Epilogue: C/D layout m = r + 8*half, n = laneN
#pragma unroll
  for (int mi = 0; mi < 4; mi++)
#pragma unroll
    for (int ni = 0; ni < 4; ni++)
#pragma unroll
      for (int r = 0; r < 8; r++) {
        int m = m0 + wm * 64 + mi * 16 + r + 8 * half;
        int n = n0 + wn * 64 + ni * 16 + laneN;
        if (OUT_BF16)
          ((uint16_t*)Cout)[(size_t)m * Nn + n] = f2b(acc[mi][ni][r]);
        else
          ((float*)Cout)[(size_t)m * Nn + n] = acc[mi][ni][r];
      }
}

// --------------------------------------------------------- flash attention ---
// Grid: (B*H, N/128). Block 256 = 8 waves; wave w owns 16 query rows.
// Keys in tiles of 32, double-buffered K/V staging, online softmax in exp2
// domain; row sums come free from a "ones column" WMMA accumulator (o[4]).
__global__ __launch_bounds__(256) void attention_kernel(
    const uint16_t* __restrict__ qkv,      // [B, N, 3D] bf16
    uint16_t* __restrict__ obuf) {         // [B, N, D]  bf16 (heads merged)
  __shared__ __align__(16) uint16_t Ks[2][32 * 72];     // [key][dk], pad 72
  __shared__ __align__(16) uint16_t Vs[2][64 * 40];     // transposed [dk][key]
  __shared__ __align__(16) uint16_t Pb[8 * 16 * 40];    // per-wave P repack

  const int tid   = threadIdx.x;
  const int lane  = tid & 31;
  const int wid   = tid >> 5;          // 0..7
  const int laneN = lane & 15;
  const int half  = lane >> 4;
  const int bh = blockIdx.x;
  const int b = bh / HH, h = bh % HH;
  const int qbase = blockIdx.y * 128;
  const size_t rs = 3 * DD;

  const uint16_t* qp = qkv + (size_t)b * NN * rs + h * DK;
  const uint16_t* kp = qp + DD;
  const uint16_t* vp = qp + 2 * DD;

  auto stageK = [&](int kt, int bi) {
    int key = tid >> 3, seg = tid & 7;       // 0..31, 0..7
    async_copy16(kp + (size_t)(kt + key) * rs + seg * 8,
                 &Ks[bi][key * 72 + seg * 8]);
  };
  auto stageV = [&](int kt, int bi) {
    int kpair = tid >> 4, seg = tid & 15;    // keys 2kp,2kp+1; dk seg of 4
    const uint16_t* g0 = vp + (size_t)(kt + 2 * kpair) * rs + seg * 4;
    uint2 a2 = *(const uint2*)g0;
    uint2 b2 = *(const uint2*)(g0 + rs);
    const uint32_t* aw = (const uint32_t*)&a2;
    const uint32_t* bw = (const uint32_t*)&b2;
#pragma unroll
    for (int j = 0; j < 4; j++) {
      uint32_t av = (aw[j >> 1] >> ((j & 1) * 16)) & 0xFFFFu;
      uint32_t bv = (bw[j >> 1] >> ((j & 1) * 16)) & 0xFFFFu;
      *(uint32_t*)&Vs[bi][(seg * 4 + j) * 40 + 2 * kpair] = av | (bv << 16);
    }
  };

  // Q fragments (A 16x32 layout), dk halves 0..31 / 32..63
  V16 qf[2];
  {
    int qrow = qbase + wid * 16 + laneN;
    const uint16_t* qr = qp + (size_t)qrow * rs;
#pragma unroll
    for (int f = 0; f < 2; f++)
#pragma unroll
      for (int j = 0; j < 8; j++) {
        int kk = f * 32 + 2 * j + (j >= 4 ? 8 : 0) + 8 * half;
        qf[f].u[j] = *(const uint32_t*)(qr + kk);
      }
  }

  // Ones-column B fragment: column 0 of the extra V tile is 1.0 -> o[4]
  // accumulates exact softmax row sums under the same alpha recurrence.
  V16 onesb;
#pragma unroll
  for (int j = 0; j < 8; j++) onesb.u[j] = (laneN == 0) ? 0x3F803F80u : 0u;

  float mrow[8];
  v8f o[5], zero8;
#pragma unroll
  for (int r = 0; r < 8; r++) { mrow[r] = -1e30f; zero8[r] = 0.f; }
#pragma unroll
  for (int t = 0; t < 5; t++)
#pragma unroll
    for (int r = 0; r < 8; r++) o[t][r] = 0.f;

  const float c2 = 0.125f * 1.44269504089f;   // (1/sqrt(DK)) * log2(e)
  uint16_t* Pw = Pb + wid * 16 * 40;

  stageK(0, 0);
  stageV(0, 0);

  for (int it = 0; it < NN / 32; ++it) {
    const int kt = it * 32;
    const int cur = it & 1;
    async_wait0();
    __syncthreads();
    if (kt + 32 < NN) {
      stageK(kt + 32, cur ^ 1);
      stageV(kt + 32, cur ^ 1);
      if (kt + 64 < NN) {
        __builtin_prefetch(kp + (size_t)(kt + 64 + (tid & 31)) * rs, 0, 1);
        __builtin_prefetch(vp + (size_t)(kt + 64 + (tid & 31)) * rs, 0, 1);
      }
    }

    // Load all four K fragments (distinct regs), then interleave the two
    // S-accumulator WMMA chains to hide D->C latency and batch ds waits.
    V16 kb[2][2];                      // [ksub][dk-half]
#pragma unroll
    for (int ks = 0; ks < 2; ks++)
#pragma unroll
      for (int f = 0; f < 2; f++)
#pragma unroll
        for (int j = 0; j < 8; j++) {
          int kk = f * 32 + 2 * j + (j >= 4 ? 8 : 0) + 8 * half;
          kb[ks][f].u[j] =
              *(const uint32_t*)&Ks[cur][(ks * 16 + laneN) * 72 + kk];
        }
    v8f s0, s1;
    s0 = __builtin_amdgcn_wmma_f32_16x16x32_bf16(false, qf[0].v, false, kb[0][0].v,
                                                 (short)0, zero8, false, false);
    s1 = __builtin_amdgcn_wmma_f32_16x16x32_bf16(false, qf[0].v, false, kb[1][0].v,
                                                 (short)0, zero8, false, false);
    s0 = __builtin_amdgcn_wmma_f32_16x16x32_bf16(false, qf[1].v, false, kb[0][1].v,
                                                 (short)0, s0, false, false);
    s1 = __builtin_amdgcn_wmma_f32_16x16x32_bf16(false, qf[1].v, false, kb[1][1].v,
                                                 (short)0, s1, false, false);

    // Online softmax in exp2 domain; max via DPP row reduction
#pragma unroll
    for (int r = 0; r < 8; r++) {
      float a0 = s0[r] * c2;
      float a1 = s1[r] * c2;
      float mn = rowmax16(fmaxf(a0, a1));
      mn = fmaxf(mrow[r], mn);
      float alpha = exp2f(mrow[r] - mn);
      mrow[r] = mn;
#pragma unroll
      for (int t = 0; t < 5; t++) o[t][r] *= alpha;
      s0[r] = exp2f(a0 - mn);
      s1[r] = exp2f(a1 - mn);
    }

    // Repack P (C layout -> A layout) through per-wave LDS
#pragma unroll
    for (int r = 0; r < 8; r++) {
      Pw[(r + 8 * half) * 40 + laneN]      = f2b(s0[r]);
      Pw[(r + 8 * half) * 40 + 16 + laneN] = f2b(s1[r]);
    }
    asm volatile("s_wait_dscnt 0" ::: "memory");

    V16 pf;
#pragma unroll
    for (int j = 0; j < 8; j++) {
      int kk = 2 * j + (j >= 4 ? 8 : 0) + 8 * half;
      pf.u[j] = *(const uint32_t*)&Pw[laneN * 40 + kk];
    }

    // Load all V fragments, then O += P*V (4 tiles) and o[4] += P*ones
    V16 vb[4];
#pragma unroll
    for (int t = 0; t < 4; t++)
#pragma unroll
      for (int j = 0; j < 8; j++) {
        int kk = 2 * j + (j >= 4 ? 8 : 0) + 8 * half;
        vb[t].u[j] = *(const uint32_t*)&Vs[cur][(t * 16 + laneN) * 40 + kk];
      }
#pragma unroll
    for (int t = 0; t < 4; t++)
      o[t] = __builtin_amdgcn_wmma_f32_16x16x32_bf16(
          false, pf.v, false, vb[t].v, (short)0, o[t], false, false);
    o[4] = __builtin_amdgcn_wmma_f32_16x16x32_bf16(
        false, pf.v, false, onesb.v, (short)0, o[4], false, false);
  }

  // Normalize by row sums (column 0 of o[4]) and write merged-head O
#pragma unroll
  for (int r = 0; r < 8; r++) {
    float l = __shfl(o[4][r], half * 16, 32);   // broadcast lane n==0 of group
    float inv = 1.0f / l;
    int qrow = qbase + wid * 16 + r + 8 * half;
    size_t base = ((size_t)b * NN + qrow) * DD + h * DK;
#pragma unroll
    for (int t = 0; t < 4; t++)
      obuf[base + t * 16 + laneN] = f2b(o[t][r] * inv);
  }
}

// ---------------------------------------------------------------- launcher ---
extern "C" void kernel_launch(void* const* d_in, const int* in_sizes, int n_in,
                              void* d_out, int out_size, void* d_ws, size_t ws_size,
                              hipStream_t stream) {
  const float* x    = (const float*)d_in[0];
  const float* wqkv = (const float*)d_in[1];
  const float* wout = (const float*)d_in[2];
  float* out = (float*)d_out;

  char* ws = (char*)d_ws;
  size_t off = 0;
  auto carve = [&](size_t elems) {
    uint16_t* p = (uint16_t*)(ws + off);
    off += ((elems * sizeof(uint16_t)) + 255) & ~(size_t)255;
    return p;
  };
  uint16_t* xb    = carve((size_t)BB * NN * DD);        // 16 MB
  uint16_t* wqkvb = carve((size_t)DD * 3 * DD);         // 6 MB
  uint16_t* woutb = carve((size_t)DD * DD);             // 2 MB
  uint16_t* qkvb  = carve((size_t)BB * NN * 3 * DD);    // 48 MB
  uint16_t* ob    = carve((size_t)BB * NN * DD);        // 16 MB
  (void)ws_size; (void)in_sizes; (void)n_in; (void)out_size;

  // 1) f32 -> bf16 conversions
  cvt_f32_bf16<<<4096, 256, 0, stream>>>(x, xb, BB * NN * DD);
  cvt_f32_bf16<<<2048, 256, 0, stream>>>(wqkv, wqkvb, DD * 3 * DD);
  cvt_f32_bf16<<<1024, 256, 0, stream>>>(wout, woutb, DD * DD);

  // 2) QKV projection: [8192,1024] x [1024,3072] -> bf16
  gemm_bf16<true><<<dim3(3 * DD / 256, BB * NN / 128), 256, 0, stream>>>(
      xb, wqkvb, qkvb, BB * NN, 3 * DD, DD);

  // 3) Flash attention per (b,h, 128-query tile)
  attention_kernel<<<dim3(BB * HH, NN / 128), 256, 0, stream>>>(qkvb, ob);

  // 4) Output projection: [8192,1024] x [1024,1024] -> f32
  gemm_bf16<false><<<dim3(DD / 256, BB * NN / 128), 256, 0, stream>>>(
      ob, woutb, out, BB * NN, DD, DD);
}
